// TCFormerDynamicBlock_28063316312346
// MI455X (gfx1250) — matrix operand order repaired and checked by reading, not compiled, not run on hardware
//
#include <hip/hip_runtime.h>
#include <hip/hip_bf16.h>
#include <math.h>

// ---------------------------------------------------------------------------
// TCFormer dynamic block for MI455X (gfx1250, wave32, WMMA 16x16x32 f16)
// Fixed problem size from setup_inputs():
//   B=8, N=1568, C=256, H1=1024, H=W=56, H_init=W_init=112, heads=8, d=32
//   M = 28*28 = 784 keys (sr=2), padded to 800 for 32-key attention tiles.
// ---------------------------------------------------------------------------

typedef _Float16 v16h __attribute__((ext_vector_type(16)));
typedef _Float16 v8h  __attribute__((ext_vector_type(8)));
typedef float    v8f  __attribute__((ext_vector_type(8)));

union Frag { v16h v; v8h h[2]; };

#define B_   8
#define N_   1568
#define C_   256
#define H1_  1024
#define HW_  56
#define M_   784
#define MP_  800
#define NI_  12544      /* H_init*W_init = 112*112 */
#define BN_  12544      /* B*N */
#define BM_  6272       /* B*M */
#define W4_  (1.0f/(4.0f+1e-6f))

__device__ __forceinline__ v8f wmma_f16(v16h a, v16h b, v8f c) {
  return __builtin_amdgcn_wmma_f32_16x16x32_f16(false, a, false, b, (short)0, c,
                                                false, false);
}

// ------------------------------ LayerNorm ----------------------------------
// one block (256 threads) per row of 256, writes f16
__global__ __launch_bounds__(256) void layernorm_k(
    const float* __restrict__ in, const float* __restrict__ w,
    const float* __restrict__ bb, _Float16* __restrict__ outH)
{
  __shared__ float red[256];
  int row = blockIdx.x;
  int t = threadIdx.x;
  float x = in[(size_t)row * C_ + t];
  red[t] = x; __syncthreads();
  #pragma unroll
  for (int s = 128; s > 0; s >>= 1) {
    if (t < s) red[t] += red[t + s];
    __syncthreads();
  }
  float mean = red[0] * (1.0f / C_);
  __syncthreads();
  float xc = x - mean;
  red[t] = xc * xc; __syncthreads();
  #pragma unroll
  for (int s = 128; s > 0; s >>= 1) {
    if (t < s) red[t] += red[t + s];
    __syncthreads();
  }
  float var = red[0] * (1.0f / C_);
  float y = xc * rsqrtf(var + 1e-5f) * w[t] + bb[t];
  outH[(size_t)row * C_ + t] = (_Float16)y;
}

// --------------------------- B-matrix packing ------------------------------
// packs W (K x N, f32 row-major) into per-lane WMMA B fragments (f16):
//   P[((kt*ntiles + nt)*32 + lane)*16 + j] = W[kt*32 + (lane<16?0:16)+j][nt*16 + (lane&15)]
__global__ void pack_b(const float* __restrict__ W, _Float16* __restrict__ P,
                       int K, int N)
{
  int tid = blockIdx.x * blockDim.x + threadIdx.x;
  if (tid >= K * N) return;
  int j = tid & 15;
  int lane = (tid >> 4) & 31;
  int rest = tid >> 9;
  int ntiles = N >> 4;
  int nt = rest % ntiles;
  int kt = rest / ntiles;
  int k = kt * 32 + ((lane < 16) ? 0 : 16) + j;
  int n = nt * 16 + (lane & 15);
  P[tid] = (_Float16)W[(size_t)k * N + n];
}

// ------------------------------ WMMA GEMM ----------------------------------
// one wave computes a 64x16 strip of D (4 M-tiles sharing each B fragment);
// block = 4 waves covering 64x64. A sched_barrier pins all fragment loads
// before the WMMA phase so waits retire progressively (no full-drain per MMA).
// modes: 0 = f32 out (+optional residual), 1 = q heads f16 (*scale),
//        2 = kv split -> k heads (padded MP_) + v transposed
__global__ __launch_bounds__(128) void gemm16(
    const _Float16* __restrict__ A, const _Float16* __restrict__ Bp,
    const float* __restrict__ bias, const float* __restrict__ resid,
    float* __restrict__ outF, _Float16* __restrict__ out1,
    _Float16* __restrict__ out2, int M, int N, int K, int mode,
    int rowsPerBatch, float scale)
{
  int lane = threadIdx.x & 31;
  int wv   = threadIdx.x >> 5;
  int m0   = blockIdx.x * 64;
  int nt   = blockIdx.y * 4 + wv;
  int n0   = nt * 16;
  int rowb = m0 + (lane & 15);
  int cb   = (lane < 16) ? 0 : 16;
  int nTiles = N >> 4;
  v8f acc[4];
  #pragma unroll
  for (int mi = 0; mi < 4; ++mi)
    acc[mi] = (v8f){0.f,0.f,0.f,0.f,0.f,0.f,0.f,0.f};
  for (int k0 = 0; k0 < K; k0 += 32) {
    Frag bf, af[4];
    const _Float16* bp = Bp + ((((size_t)(k0 >> 5) * nTiles + nt) * 32 + lane) << 4);
    bf.h[0] = *(const v8h*)(bp);
    bf.h[1] = *(const v8h*)(bp + 8);
    __builtin_prefetch(bp + (size_t)nTiles * 512, 0, 1);  // next k-tile of B
    #pragma unroll
    for (int mi = 0; mi < 4; ++mi) {
      const _Float16* ap = A + (size_t)(rowb + mi * 16) * K + k0 + cb;
      af[mi].h[0] = *(const v8h*)(ap);
      af[mi].h[1] = *(const v8h*)(ap + 8);
    }
    // keep all 10 loads issued ahead of the MMA phase
    __builtin_amdgcn_sched_barrier(0);
    #pragma unroll
    for (int mi = 0; mi < 4; ++mi)
      acc[mi] = wmma_f16(af[mi].v, bf.v, acc[mi]);
  }
  int col16 = lane & 15;
  int rbase = (lane < 16) ? 0 : 8;
  #pragma unroll
  for (int mi = 0; mi < 4; ++mi) {
    #pragma unroll
    for (int r = 0; r < 8; ++r) {
      int rr = m0 + mi * 16 + rbase + r;
      int cc = n0 + col16;
      float v = acc[mi][r] + bias[cc];
      if (mode == 0) {
        if (resid) v += resid[(size_t)rr * N + cc];
        outF[(size_t)rr * N + cc] = v;
      } else if (mode == 1) {
        v *= scale;
        int b = rr / rowsPerBatch, n = rr % rowsPerBatch;
        int h = cc >> 5, dd = cc & 31;
        out1[(((size_t)(b * 8 + h) * rowsPerBatch + n) << 5) + dd] = (_Float16)v;
      } else { // mode 2: kv split
        int b = rr / rowsPerBatch, mrow = rr % rowsPerBatch;
        int pair = cc >> 8, h = (cc >> 5) & 7, dd = cc & 31;
        if (pair == 0)
          out1[(((size_t)(b * 8 + h) * MP_ + mrow) << 5) + dd] = (_Float16)v;
        else
          out2[((size_t)(b * 8 + h) * 32 + dd) * MP_ + mrow] = (_Float16)v;
      }
    }
  }
}

// ---------------------- token2map -> im2col for 2x2/s2 conv ----------------
// counts==4 structurally (H_init=2H), so token2map is a 4-token gather.
// one block per (b, output cell m in 28x28); 256 threads = input channels.
__global__ __launch_bounds__(256) void build_im2col(
    const _Float16* __restrict__ xn, const float* __restrict__ token_score,
    const int* __restrict__ idx_token, _Float16* __restrict__ im2col,
    float* __restrict__ confp)
{
  int cell = blockIdx.x;              // b*784 + m
  int b = cell / M_, m = cell % M_;
  int rr = m / 28, cc = m % 28;
  int cin = threadIdx.x;
  const int* idx = idx_token + (size_t)b * NI_;
  for (int s = 0; s < 4; ++s) {
    int r = rr * 2 + (s >> 1), c = cc * 2 + (s & 1);   // 56x56 map cell
    float sum = 0.f;
    #pragma unroll
    for (int a = 0; a < 2; ++a)
      #pragma unroll
      for (int q = 0; q < 2; ++q) {
        int tok = idx[(r * 2 + a) * 112 + (c * 2 + q)];
        sum += (float)xn[((size_t)(b * N_ + tok)) * C_ + cin];
      }
    im2col[(size_t)cell * 1024 + s * C_ + cin] = (_Float16)(sum * W4_);
  }
  if (cin == 0) {
    float sc = 0.f;
    for (int s = 0; s < 4; ++s) {
      int r = rr * 2 + (s >> 1), c = cc * 2 + (s & 1);
      for (int a = 0; a < 2; ++a)
        for (int q = 0; q < 2; ++q) {
          int tok = idx[(r * 2 + a) * 112 + (c * 2 + q)];
          sc += token_score[(size_t)b * N_ + tok];
        }
    }
    confp[b * MP_ + m] = sc * W4_ * 0.25f;
  }
}

__global__ void pad_conf(float* __restrict__ confp)
{
  int t = blockIdx.x * blockDim.x + threadIdx.x;
  if (t >= B_ * (MP_ - M_)) return;
  int b = t / (MP_ - M_), m = M_ + t % (MP_ - M_);
  confp[b * MP_ + m] = -1e30f;
}

// --------------------------- Flash attention -------------------------------
// one wave per (b*8+h, 16-query tile). S = q k^T + conf, online softmax,
// O += P V via WMMA; V pre-transposed for contiguous B fragments.
__global__ __launch_bounds__(32) void attention_k(
    const _Float16* __restrict__ qh, const _Float16* __restrict__ kh,
    const _Float16* __restrict__ vt, const float* __restrict__ confp,
    _Float16* __restrict__ aout)
{
  __shared__ __align__(16) _Float16 pl[16][40];
  int lane = threadIdx.x;
  int bh = blockIdx.x;                 // b*8 + head
  int b = bh >> 3, hd = bh & 7;
  int n0 = blockIdx.y * 16;
  int row16 = lane & 15;
  int hi = (lane < 16) ? 0 : 1;
  Frag qa;
  {
    const _Float16* qp = qh + (((size_t)bh * N_ + n0 + row16) << 5) + (hi ? 16 : 0);
    qa.h[0] = *(const v8h*)(qp);
    qa.h[1] = *(const v8h*)(qp + 8);
  }
  float mst[8], lst[8];
  v8f acc0 = {0.f,0.f,0.f,0.f,0.f,0.f,0.f,0.f};
  v8f acc1 = {0.f,0.f,0.f,0.f,0.f,0.f,0.f,0.f};
  #pragma unroll
  for (int r = 0; r < 8; ++r) { mst[r] = -1e30f; lst[r] = 0.f; }
  const float* cf = confp + b * MP_;
  for (int it = 0; it < MP_ / 32; ++it) {
    int k0 = it * 32;
    // load both K fragments, then both S WMMAs
    Frag kb[2];
    #pragma unroll
    for (int t = 0; t < 2; ++t) {
      const _Float16* kp =
          kh + (((size_t)bh * MP_ + k0 + t * 16 + row16) << 5) + (hi ? 16 : 0);
      kb[t].h[0] = *(const v8h*)(kp);
      kb[t].h[1] = *(const v8h*)(kp + 8);
    }
    __builtin_amdgcn_sched_barrier(0);
    v8f s[2];
    #pragma unroll
    for (int t = 0; t < 2; ++t) {
      v8f z = {0.f,0.f,0.f,0.f,0.f,0.f,0.f,0.f};
      s[t] = wmma_f16(qa.v, kb[t].v, z);
      float cv = cf[k0 + t * 16 + row16];
      #pragma unroll
      for (int r = 0; r < 8; ++r) s[t][r] += cv;
    }
    float mnew[8], alpha[8];
    #pragma unroll
    for (int r = 0; r < 8; ++r) {
      float mx = fmaxf(s[0][r], s[1][r]);
      #pragma unroll
      for (int d = 1; d < 16; d <<= 1) mx = fmaxf(mx, __shfl_xor(mx, d, 32));
      mnew[r] = fmaxf(mst[r], mx);
      alpha[r] = __expf(mst[r] - mnew[r]);
    }
    #pragma unroll
    for (int r = 0; r < 8; ++r) {
      float p0 = __expf(s[0][r] - mnew[r]);
      float p1 = __expf(s[1][r] - mnew[r]);
      float rs = p0 + p1;
      #pragma unroll
      for (int d = 1; d < 16; d <<= 1) rs += __shfl_xor(rs, d, 32);
      lst[r] = lst[r] * alpha[r] + rs;
      mst[r] = mnew[r];
      acc0[r] *= alpha[r];
      acc1[r] *= alpha[r];
      pl[hi * 8 + r][row16]      = (_Float16)p0;
      pl[hi * 8 + r][16 + row16] = (_Float16)p1;
    }
    __syncthreads();
    Frag pa;
    {
      const _Float16* pp = &pl[row16][hi ? 16 : 0];
      pa.h[0] = *(const v8h*)(pp);
      pa.h[1] = *(const v8h*)(pp + 8);
    }
    // load both V fragments, then both O WMMAs
    Frag vb[2];
    #pragma unroll
    for (int h2 = 0; h2 < 2; ++h2) {
      const _Float16* vp =
          vt + ((size_t)bh * 32 + h2 * 16 + row16) * MP_ + k0 + (hi ? 16 : 0);
      vb[h2].h[0] = *(const v8h*)(vp);
      vb[h2].h[1] = *(const v8h*)(vp + 8);
    }
    __builtin_amdgcn_sched_barrier(0);
    acc0 = wmma_f16(pa.v, vb[0].v, acc0);
    acc1 = wmma_f16(pa.v, vb[1].v, acc1);
    __syncthreads();
  }
  #pragma unroll
  for (int r = 0; r < 8; ++r) {
    int qrow = n0 + hi * 8 + r;
    float linv = 1.0f / lst[r];
    size_t base = ((size_t)(b * N_ + qrow)) * C_ + hd * 32;
    aout[base + row16]      = (_Float16)(acc0[r] * linv);
    aout[base + 16 + row16] = (_Float16)(acc1[r] * linv);
  }
}

// --------------------- map2token support: token histogram ------------------
__global__ void hist_counts(const int* __restrict__ idx_token,
                            int* __restrict__ counts)
{
  int i = blockIdx.x * blockDim.x + threadIdx.x;
  if (i >= B_ * NI_) return;
  int b = i / NI_;
  atomicAdd(&counts[b * N_ + idx_token[i]], 1);
}

// --------------------- token2map of h (1024 ch) -> hmap --------------------
__global__ __launch_bounds__(256) void h_token2map(
    const float* __restrict__ h, const int* __restrict__ idx_token,
    float* __restrict__ hmap)
{
  int cell = blockIdx.x;               // b*3136 + r*56 + c
  int b = cell / (HW_ * HW_), rc = cell % (HW_ * HW_);
  int r = rc / HW_, c = rc % HW_;
  const int* idx = idx_token + (size_t)b * NI_;
  int toks[4];
  #pragma unroll
  for (int a = 0; a < 2; ++a)
    #pragma unroll
    for (int q = 0; q < 2; ++q)
      toks[a * 2 + q] = idx[(r * 2 + a) * 112 + (c * 2 + q)];
  for (int j = 0; j < 4; ++j) {
    int ch = threadIdx.x + j * 256;
    float sum = 0.f;
    #pragma unroll
    for (int s = 0; s < 4; ++s)
      sum += h[((size_t)(b * N_ + toks[s])) * H1_ + ch];
    hmap[(size_t)cell * H1_ + ch] = sum * W4_;
  }
}

// --------------- fused 3x3 depthwise conv + map2token scatter --------------
__global__ __launch_bounds__(256) void dwconv_map2token(
    const float* __restrict__ hmap, const float* __restrict__ dw_w,
    const float* __restrict__ dw_b, const int* __restrict__ idx_token,
    const int* __restrict__ counts, float* __restrict__ mtok)
{
  int cell = blockIdx.x;
  int b = cell / (HW_ * HW_), rc = cell % (HW_ * HW_);
  int r = rc / HW_, c = rc % HW_;
  const int* idx = idx_token + (size_t)b * NI_;
  int toks[4]; float wts[4];
  #pragma unroll
  for (int a = 0; a < 2; ++a)
    #pragma unroll
    for (int q = 0; q < 2; ++q) {
      int tok = idx[(r * 2 + a) * 112 + (c * 2 + q)];
      toks[a * 2 + q] = tok;
      wts[a * 2 + q] = 1.0f / ((float)counts[b * N_ + tok] + 1e-6f);
    }
  for (int j = 0; j < 4; ++j) {
    int ch = threadIdx.x + j * 256;
    float acc = dw_b[ch];
    #pragma unroll
    for (int dr = -1; dr <= 1; ++dr)
      #pragma unroll
      for (int dc = -1; dc <= 1; ++dc) {
        int rr = r + dr, c2 = c + dc;
        if (rr >= 0 && rr < HW_ && c2 >= 0 && c2 < HW_)
          acc += hmap[((size_t)(b * HW_ * HW_) + rr * HW_ + c2) * H1_ + ch] *
                 dw_w[((dr + 1) * 3 + (dc + 1)) * H1_ + ch];
      }
    #pragma unroll
    for (int s = 0; s < 4; ++s)
      atomicAdd(&mtok[((size_t)(b * N_ + toks[s])) * H1_ + ch], acc * wts[s]);
  }
}

// ----------------------- hc = gelu(h*skip + mtok) --------------------------
__global__ void hc_gelu(const float* __restrict__ h,
                        const float* __restrict__ skip_w,
                        const float* __restrict__ mtok,
                        _Float16* __restrict__ hc)
{
  size_t i = (size_t)blockIdx.x * blockDim.x + threadIdx.x;
  if (i >= (size_t)BN_ * H1_) return;
  int ch = (int)(i & (H1_ - 1));
  float v = h[i] * skip_w[ch] + mtok[i];
  float g = 0.5f * v * (1.0f + erff(v * 0.70710678118654752f));
  hc[i] = (_Float16)g;
}

// ===========================================================================
extern "C" void kernel_launch(void* const* d_in, const int* in_sizes, int n_in,
                              void* d_out, int out_size, void* d_ws,
                              size_t ws_size, hipStream_t stream)
{
  (void)in_sizes; (void)n_in; (void)out_size; (void)ws_size;
  const float* x    = (const float*)d_in[0];
  const float* tsc  = (const float*)d_in[1];
  const int*   idxt = (const int*)d_in[2];
  const float* n1w  = (const float*)d_in[3];
  const float* n1b  = (const float*)d_in[4];
  const float* qw   = (const float*)d_in[5];
  const float* qb   = (const float*)d_in[6];
  const float* kvw  = (const float*)d_in[7];
  const float* kvb  = (const float*)d_in[8];
  const float* srw  = (const float*)d_in[9];
  const float* srb  = (const float*)d_in[10];
  const float* srnw = (const float*)d_in[11];
  const float* srnb = (const float*)d_in[12];
  const float* pw   = (const float*)d_in[13];
  const float* pb   = (const float*)d_in[14];
  const float* n2w  = (const float*)d_in[15];
  const float* n2b  = (const float*)d_in[16];
  const float* f1w  = (const float*)d_in[17];
  const float* f1b  = (const float*)d_in[18];
  const float* skw  = (const float*)d_in[19];
  const float* dww  = (const float*)d_in[20];
  const float* dwb  = (const float*)d_in[21];
  const float* f2w  = (const float*)d_in[22];
  const float* f2b  = (const float*)d_in[23];
  float* outp = (float*)d_out;

  char* p = (char*)d_ws;
  auto alloc = [&](size_t bytes) -> void* {
    void* r = (void*)p;
    p += (bytes + 255) & ~(size_t)255;
    return r;
  };
  _Float16* xn     = (_Float16*)alloc((size_t)BN_ * C_ * 2);
  _Float16* qh     = (_Float16*)alloc((size_t)BN_ * C_ * 2);
  _Float16* im2c   = (_Float16*)alloc((size_t)BM_ * 1024 * 2);
  float*    confp  = (float*)   alloc((size_t)B_ * MP_ * 4);
  float*    kvconv = (float*)   alloc((size_t)BM_ * C_ * 4);
  _Float16* kvn    = (_Float16*)alloc((size_t)BM_ * C_ * 2);
  _Float16* kf     = (_Float16*)alloc((size_t)64 * MP_ * 32 * 2);
  _Float16* vt     = (_Float16*)alloc((size_t)64 * MP_ * 32 * 2);
  _Float16* af16   = (_Float16*)alloc((size_t)BN_ * C_ * 2);
  float*    x2     = (float*)   alloc((size_t)BN_ * C_ * 4);
  _Float16* h2n    = (_Float16*)alloc((size_t)BN_ * C_ * 2);
  float*    hbuf   = (float*)   alloc((size_t)BN_ * H1_ * 4);
  float*    hmap   = (float*)   alloc((size_t)B_ * HW_ * HW_ * H1_ * 4);
  float*    mtok   = (float*)   alloc((size_t)BN_ * H1_ * 4);
  _Float16* hcf    = (_Float16*)alloc((size_t)BN_ * H1_ * 2);
  int*      counts = (int*)     alloc((size_t)B_ * N_ * 4);
  _Float16* qp     = (_Float16*)alloc((size_t)256 * 256 * 2);
  _Float16* kvp    = (_Float16*)alloc((size_t)256 * 512 * 2);
  _Float16* srp    = (_Float16*)alloc((size_t)1024 * 256 * 2);
  _Float16* projp  = (_Float16*)alloc((size_t)256 * 256 * 2);
  _Float16* f1p    = (_Float16*)alloc((size_t)256 * 1024 * 2);
  _Float16* f2p    = (_Float16*)alloc((size_t)1024 * 256 * 2);

  // weight packing
  pack_b<<<(256 * 256 + 255) / 256, 256, 0, stream>>>(qw, qp, 256, 256);
  pack_b<<<(256 * 512 + 255) / 256, 256, 0, stream>>>(kvw, kvp, 256, 512);
  pack_b<<<(1024 * 256 + 255) / 256, 256, 0, stream>>>(srw, srp, 1024, 256);
  pack_b<<<(256 * 256 + 255) / 256, 256, 0, stream>>>(pw, projp, 256, 256);
  pack_b<<<(256 * 1024 + 255) / 256, 256, 0, stream>>>(f1w, f1p, 256, 1024);
  pack_b<<<(1024 * 256 + 255) / 256, 256, 0, stream>>>(f2w, f2p, 1024, 256);

  // zero-init (k/v pad rows, scatter accumulator, histogram)
  hipMemsetAsync(kf, 0, (size_t)64 * MP_ * 32 * 2, stream);
  hipMemsetAsync(vt, 0, (size_t)64 * MP_ * 32 * 2, stream);
  hipMemsetAsync(mtok, 0, (size_t)BN_ * H1_ * 4, stream);
  hipMemsetAsync(counts, 0, (size_t)B_ * N_ * 4, stream);

  const float scale = 0.17677669529663687f; // 1/sqrt(32)

  // 1) LN1
  layernorm_k<<<BN_, 256, 0, stream>>>(x, n1w, n1b, xn);
  // 2) q = xn @ q_w + q_b  (f16 heads, *scale)
  gemm16<<<dim3(BN_ / 64, 256 / 64), 128, 0, stream>>>(
      xn, qp, qb, nullptr, nullptr, qh, nullptr, BN_, 256, 256, 1, N_, scale);
  // 3) token2map + im2col for 2x2/s2 conv, + conf pooling
  build_im2col<<<BM_, 256, 0, stream>>>(xn, tsc, idxt, im2c, confp);
  pad_conf<<<1, 128, 0, stream>>>(confp);
  // 4) sr conv as GEMM: [6272,1024] x [1024,256]
  gemm16<<<dim3(BM_ / 64, 256 / 64), 128, 0, stream>>>(
      im2c, srp, srb, nullptr, kvconv, nullptr, nullptr, BM_, 256, 1024, 0, M_, 1.f);
  // 5) LN (srn)
  layernorm_k<<<BM_, 256, 0, stream>>>(kvconv, srnw, srnb, kvn);
  // 6) kv GEMM: split into k heads (padded) + v transposed
  gemm16<<<dim3(BM_ / 64, 512 / 64), 128, 0, stream>>>(
      kvn, kvp, kvb, nullptr, nullptr, kf, vt, BM_, 512, 256, 2, M_, 1.f);
  // 7) flash attention
  attention_k<<<dim3(64, N_ / 16), 32, 0, stream>>>(qh, kf, vt, confp, af16);
  // 8) proj + residual(x) -> x2
  gemm16<<<dim3(BN_ / 64, 256 / 64), 128, 0, stream>>>(
      af16, projp, pb, x, x2, nullptr, nullptr, BN_, 256, 256, 0, N_, 1.f);
  // 9) LN2
  layernorm_k<<<BN_, 256, 0, stream>>>(x2, n2w, n2b, h2n);
  // 10) fc1 -> h (f32)
  gemm16<<<dim3(BN_ / 64, 1024 / 64), 128, 0, stream>>>(
      h2n, f1p, f1b, nullptr, hbuf, nullptr, nullptr, BN_, 1024, 256, 0, N_, 1.f);
  // 11) map2token prep + token2map(h) + fused dwconv+scatter
  hist_counts<<<(B_ * NI_ + 255) / 256, 256, 0, stream>>>(idxt, counts);
  h_token2map<<<B_ * HW_ * HW_, 256, 0, stream>>>(hbuf, idxt, hmap);
  dwconv_map2token<<<B_ * HW_ * HW_, 256, 0, stream>>>(hmap, dww, dwb, idxt,
                                                       counts, mtok);
  // 12) hc = gelu(h*skip + mtok)
  hc_gelu<<<(unsigned)(((size_t)BN_ * H1_ + 255) / 256), 256, 0, stream>>>(
      hbuf, skw, mtok, hcf);
  // 13) fc2 + residual(x2) -> d_out
  gemm16<<<dim3(BN_ / 64, 256 / 64), 128, 0, stream>>>(
      hcf, f2p, f2b, x2, outp, nullptr, nullptr, BN_, 256, 1024, 0, N_, 1.f);
}
